// Attention_80779744903968
// MI455X (gfx1250) — matrix operand order repaired and checked by reading, not compiled, not run on hardware
//
#include <hip/hip_runtime.h>
#include <cstdint>

// ---------------------------------------------------------------------------
// Attention (Luong-style) on MI455X gfx1250.
// Heavy math: v_wmma_f32_16x16x32_bf16 (f32 accumulate).
// Data movement: GLOBAL_LOAD_ASYNC_TO_LDS_B128 (ASYNCcnt) when the toolchain
// declares the builtin; otherwise register-staged b128 copies.
// GEMM is "TN": A [M,K] row-major, B transposed as Bt [N,K] row-major.
// ---------------------------------------------------------------------------

typedef __attribute__((ext_vector_type(16))) __bf16 v16bf;
typedef __attribute__((ext_vector_type(8)))  float  v8f;

#define H_DIM 1024
#define S_LEN 1024
#define T_LEN 512
#define B_SZ  32
#define BT    (B_SZ * T_LEN)   // 16384

#define BM 128
#define BN 128
#define BK 32
#define LDS_STRIDE 40          // 32 + 8 halves pad; 80B rows keep 16B alignment

#if defined(__AMDGCN__) && __has_builtin(__builtin_amdgcn_global_load_async_to_lds_b128)
#define USE_ASYNC_LDS 1
#else
#define USE_ASYNC_LDS 0
#endif

__device__ __forceinline__ unsigned short f2bf(float f) {
  unsigned u = __float_as_uint(f);
  u += 0x7FFFu + ((u >> 16) & 1u);       // round-to-nearest-even
  return (unsigned short)(u >> 16);
}

#if USE_ASYNC_LDS
// builtin signature (from clang diagnostic): param1 is a pointer to a
// gcc-vector int4 in the global address space; param2 mirrors it in LDS.
typedef int v4i_vs __attribute__((vector_size(4 * sizeof(int))));
typedef v4i_vs __attribute__((address_space(1)))* g4_t;
typedef v4i_vs __attribute__((address_space(3)))* l4_t;

__device__ __forceinline__ void async_b128(const unsigned short* g, unsigned short* l) {
  // flat LDS pointers truncate to the LDS byte offset (low 32 address bits);
  // flat global pointers are numerically identical to AS1.
  __builtin_amdgcn_global_load_async_to_lds_b128(
      (g4_t)(uintptr_t)g, (l4_t)(uintptr_t)l, 0, 0);
}
__device__ __forceinline__ void wait_async() {
#if __has_builtin(__builtin_amdgcn_s_wait_asynccnt)
  __builtin_amdgcn_s_wait_asynccnt(0);
#else
  asm volatile("s_wait_asynccnt 0" ::: "memory");
#endif
}
#endif

// ---------------------------- prep kernels ---------------------------------

__global__ void k_prep_query(const float* __restrict__ q,
                             unsigned short* __restrict__ catA, int n) {
  int i = blockIdx.x * 256 + threadIdx.x;
  if (i >= n) return;
  int m = i >> 10, h = i & 1023;                    // [BT,H]
  catA[(size_t)m * 2048 + 1024 + h] = f2bf(q[i]);   // right half of [c|q]
}

__global__ void k_cast_bf16(const float* __restrict__ src,
                            unsigned short* __restrict__ dst, int n) {
  int i = blockIdx.x * 256 + threadIdx.x;
  if (i >= n) return;
  dst[i] = f2bf(src[i]);
}

__global__ void k_prep_enc(const float* __restrict__ enc,
                           unsigned short* __restrict__ encB,   // [S,B,H] bf16
                           unsigned short* __restrict__ encT,   // [B,H,S] bf16
                           int n) {
  int i = blockIdx.x * 256 + threadIdx.x;
  if (i >= n) return;
  int s = i >> 15;                 // / (B*H)
  int rem = i & 32767;
  int b = rem >> 10, h = rem & 1023;
  unsigned short v = f2bf(enc[i]);
  encB[i] = v;
  encT[((size_t)b << 20) + ((size_t)h << 10) + s] = v;
}

// ------------------------- bf16 WMMA GEMM (TN) -----------------------------
// C[M,N](f32 acc) = A[M,K] * Bt[N,K]^T, A/Bt bf16 row-major in global.
// mode 0: store f32   mode 1: store bf16   mode 2: store f32 tanh(x + bias[n])

union FragBF { uint4 q[2]; v16bf v; };

__global__ __launch_bounds__(256)
void k_gemm(const unsigned short* __restrict__ A, long long lda, long long sA,
            const unsigned short* __restrict__ Bt, long long ldb, long long sB,
            void* __restrict__ Cv, long long ldc, long long sC,
            int K, int mode, const float* __restrict__ bias)
{
  // double-buffered tiles: 40 KB of 320 KB/WGP
  __shared__ __align__(16) unsigned short lA[2][BM * LDS_STRIDE];
  __shared__ __align__(16) unsigned short lB[2][BN * LDS_STRIDE];

  const int tid  = threadIdx.x;
  const int lane = tid & 31;
  const int wave = tid >> 5;
  const int wm = wave >> 1;              // 0..3 -> 32-row slabs
  const int wn = wave & 1;               // 0..1 -> 64-col slabs
  const int bm0 = blockIdx.y * BM;
  const int bn0 = blockIdx.x * BN;

  const unsigned short* Abase = A  + (size_t)blockIdx.z * sA + (size_t)bm0 * lda;
  const unsigned short* Bbase = Bt + (size_t)blockIdx.z * sB + (size_t)bn0 * ldb;

  v8f acc[2][4];
  #pragma unroll
  for (int i = 0; i < 2; ++i)
    #pragma unroll
    for (int j = 0; j < 4; ++j) { v8f z = {}; acc[i][j] = z; }

  const int KT = K / BK;

  // tile = 128 rows x 32 halves = 512 b128 chunks, 2 per thread
  const int c0row = tid >> 2, c0c8 = (tid & 3) * 8;            // chunk tid
  const int c1row = (tid + 256) >> 2, c1c8 = ((tid + 256) & 3) * 8;

#if USE_ASYNC_LDS
  auto asyncTile = [&](const unsigned short* gb, long long ld, int kt,
                       unsigned short* lb) {
    async_b128(gb + (size_t)c0row * ld + (size_t)kt * BK + c0c8,
               lb + c0row * LDS_STRIDE + c0c8);
    async_b128(gb + (size_t)c1row * ld + (size_t)kt * BK + c1c8,
               lb + c1row * LDS_STRIDE + c1c8);
  };
  asyncTile(Abase, lda, 0, lA[0]);
  asyncTile(Bbase, ldb, 0, lB[0]);
  wait_async();
  __syncthreads();
#else
  auto loadT = [&](const unsigned short* gb, long long ld, int kt, uint4* r) {
    r[0] = *reinterpret_cast<const uint4*>(gb + (size_t)c0row * ld + (size_t)kt * BK + c0c8);
    r[1] = *reinterpret_cast<const uint4*>(gb + (size_t)c1row * ld + (size_t)kt * BK + c1c8);
  };
  auto storeT = [&](unsigned short* lb, const uint4* r) {
    *reinterpret_cast<uint4*>(&lb[c0row * LDS_STRIDE + c0c8]) = r[0];
    *reinterpret_cast<uint4*>(&lb[c1row * LDS_STRIDE + c1c8]) = r[1];
  };
  uint4 na0[2], nb0[2];
  loadT(Abase, lda, 0, na0);
  loadT(Bbase, ldb, 0, nb0);
  storeT(lA[0], na0);
  storeT(lB[0], nb0);
  __syncthreads();
#endif

  const int hi = lane >> 4;        // half-wave select
  const int rl = lane & 15;
  const int cbA = hi * 8;          // A frag: lanes<16 K{0..7,16..23}; >=16 K{8..15,24..31}
  const int cbB = hi * 8;          // B frag: lanes<16 K{0..7,16..23}; >=16 K{8..15,24..31}

  for (int kt = 0; kt < KT; ++kt) {
    const int cur = kt & 1, nxt = cur ^ 1;

#if USE_ASYNC_LDS
    if (kt + 1 < KT) {                       // async-fill the other buffer
      asyncTile(Abase, lda, kt + 1, lA[nxt]);
      asyncTile(Bbase, ldb, kt + 1, lB[nxt]);
    }
#else
    uint4 na[2], nb[2];
    if (kt + 1 < KT) {                       // stage next tile in registers
      loadT(Abase, lda, kt + 1, na);
      loadT(Bbase, ldb, kt + 1, nb);
    }
#endif
    if (kt + 2 < KT) {                       // warm caches two tiles ahead
      __builtin_prefetch(Abase + (size_t)(tid >> 1) * lda + (size_t)(kt + 2) * BK, 0, 3);
      __builtin_prefetch(Bbase + (size_t)(tid >> 1) * ldb + (size_t)(kt + 2) * BK, 0, 3);
    }

    // compute on current buffer; only one B fragment live at a time
    FragBF af0, af1;
    {
      int r0 = wm * 32 + rl, r1 = wm * 32 + 16 + rl;
      af0.q[0] = *reinterpret_cast<const uint4*>(&lA[cur][r0 * LDS_STRIDE + cbA]);
      af0.q[1] = *reinterpret_cast<const uint4*>(&lA[cur][r0 * LDS_STRIDE + cbA + 16]);
      af1.q[0] = *reinterpret_cast<const uint4*>(&lA[cur][r1 * LDS_STRIDE + cbA]);
      af1.q[1] = *reinterpret_cast<const uint4*>(&lA[cur][r1 * LDS_STRIDE + cbA + 16]);
    }
    #pragma unroll
    for (int ni = 0; ni < 4; ++ni) {
      FragBF bfr;
      int rowb = wn * 64 + ni * 16 + rl;
      bfr.q[0] = *reinterpret_cast<const uint4*>(&lB[cur][rowb * LDS_STRIDE + cbB]);
      bfr.q[1] = *reinterpret_cast<const uint4*>(&lB[cur][rowb * LDS_STRIDE + cbB + 16]);
      acc[0][ni] = __builtin_amdgcn_wmma_f32_16x16x32_bf16(
          false, af0.v, false, bfr.v, (short)0, acc[0][ni], false, false);
      acc[1][ni] = __builtin_amdgcn_wmma_f32_16x16x32_bf16(
          false, af1.v, false, bfr.v, (short)0, acc[1][ni], false, false);
    }

#if USE_ASYNC_LDS
    if (kt + 1 < KT) wait_async();           // next buffer landed
#else
    if (kt + 1 < KT) {
      storeT(lA[nxt], na);
      storeT(lB[nxt], nb);
    }
#endif
    __syncthreads();                         // one barrier per k-step
  }

  // epilogue: C/D layout -> element (M = v + 8*hi, N = rl) per 16x16 tile
  #pragma unroll
  for (int mi = 0; mi < 2; ++mi)
    #pragma unroll
    for (int ni = 0; ni < 4; ++ni) {
      int col = bn0 + wn * 64 + ni * 16 + rl;
      #pragma unroll
      for (int v = 0; v < 8; ++v) {
        int row = bm0 + wm * 32 + mi * 16 + v + hi * 8;
        float x = acc[mi][ni][v];
        size_t off = (size_t)blockIdx.z * sC + (size_t)row * ldc + col;
        if (mode == 0)      ((float*)Cv)[off] = x;
        else if (mode == 1) ((unsigned short*)Cv)[off] = f2bf(x);
        else                ((float*)Cv)[off] = tanhf(x + bias[col]);
      }
    }
}

// ------------------------------ softmax ------------------------------------
// One block per (b,t) row: masked-fill(score==0 -> -inf), softmax over S=1024.

__global__ __launch_bounds__(256)
void k_softmax(const float* __restrict__ scores, unsigned short* __restrict__ p)
{
  __shared__ float red[256];
  const int row = blockIdx.x;
  const float* sr = scores + (size_t)row * S_LEN;
  unsigned short* pr = p + (size_t)row * S_LEN;
  const int tid = threadIdx.x;
  const float NEG = -__builtin_inff();

  float v[4], mx = NEG;
  #pragma unroll
  for (int i = 0; i < 4; ++i) {
    float x = sr[tid + i * 256];
    if (x == 0.0f) x = NEG;
    v[i] = x; mx = fmaxf(mx, x);
  }
  red[tid] = mx; __syncthreads();
  for (int s = 128; s > 0; s >>= 1) {
    if (tid < s) red[tid] = fmaxf(red[tid], red[tid + s]);
    __syncthreads();
  }
  mx = red[0]; __syncthreads();

  float sum = 0.f;
  #pragma unroll
  for (int i = 0; i < 4; ++i) { v[i] = __expf(v[i] - mx); sum += v[i]; }
  red[tid] = sum; __syncthreads();
  for (int s = 128; s > 0; s >>= 1) {
    if (tid < s) red[tid] += red[tid + s];
    __syncthreads();
  }
  float inv = 1.0f / red[0];
  #pragma unroll
  for (int i = 0; i < 4; ++i) pr[tid + i * 256] = f2bf(v[i] * inv);
}

// ------------------------------ launcher -----------------------------------

extern "C" void kernel_launch(void* const* d_in, const int* in_sizes, int n_in,
                              void* d_out, int out_size, void* d_ws, size_t ws_size,
                              hipStream_t stream)
{
  const float* q    = (const float*)d_in[0];   // [B,T,H]
  const float* enc  = (const float*)d_in[1];   // [S,B,H]
  // d_in[2] src_lengths: computed-but-unused in reference
  const float* Win  = (const float*)d_in[3];   // [H,H]   == Bt [N,K] for GEMM1
  const float* Wout = (const float*)d_in[4];   // [H,2H]  == Bt [N,K] for GEMM4
  const float* bout = (const float*)d_in[5];   // [H]
  (void)in_sizes; (void)n_in; (void)out_size; (void)ws_size;

  char* ws = (char*)d_ws;                      // ~294 MB scratch
  unsigned short* catA  = (unsigned short*)(ws);                 // [BT,2H] bf16  64MB
  unsigned short* z     = (unsigned short*)(ws +  67108864ull);  // [BT,H]  bf16  32MB
  unsigned short* WinB  = (unsigned short*)(ws + 100663296ull);  // [H,H]   bf16   2MB
  unsigned short* WoutB = (unsigned short*)(ws + 102760448ull);  // [H,2H]  bf16   4MB
  unsigned short* encB  = (unsigned short*)(ws + 106954752ull);  // [S,B,H] bf16  64MB
  unsigned short* encT  = (unsigned short*)(ws + 174063616ull);  // [B,H,S] bf16  64MB
  float*          sc    = (float*)         (ws + 241172480ull);  // [B,T,S] f32   64MB
  unsigned short* p     = z;  // alias: z fully consumed before softmax writes p

  dim3 blk(256);

  k_prep_query<<<(BT * H_DIM) / 256, blk, 0, stream>>>(q, catA, BT * H_DIM);
  k_cast_bf16 <<<(H_DIM * H_DIM) / 256, blk, 0, stream>>>(Win, WinB, H_DIM * H_DIM);
  k_cast_bf16 <<<(H_DIM * 2 * H_DIM) / 256, blk, 0, stream>>>(Wout, WoutB,
                                                              H_DIM * 2 * H_DIM);
  k_prep_enc  <<<(S_LEN * B_SZ * H_DIM) / 256, blk, 0, stream>>>(enc, encB, encT,
                                                                 S_LEN * B_SZ * H_DIM);

  // 1) z = q @ W_in^T          M=BT  N=H  K=H   (Bt = W_in itself)
  k_gemm<<<dim3(H_DIM / BN, BT / BM, 1), blk, 0, stream>>>(
      catA + H_DIM, 2 * H_DIM, 0,
      WinB, H_DIM, 0,
      z, H_DIM, 0,
      H_DIM, 1, nullptr);

  // 2) scores[b] = z[b] @ enc_b^T   M=T  N=S  K=H   (Bt = encB rows, ldb=B*H)
  k_gemm<<<dim3(S_LEN / BN, T_LEN / BM, B_SZ), blk, 0, stream>>>(
      z, H_DIM, (long long)T_LEN * H_DIM,
      encB, (long long)B_SZ * H_DIM, H_DIM,
      sc, S_LEN, (long long)T_LEN * S_LEN,
      H_DIM, 0, nullptr);

  // 3) masked-fill + softmax -> p (bf16)
  k_softmax<<<BT, blk, 0, stream>>>(sc, p);

  // 4) c[b] = p[b] @ enc_b   M=T  N=H  K=S   (Bt = encT[b])
  k_gemm<<<dim3(H_DIM / BN, T_LEN / BM, B_SZ), blk, 0, stream>>>(
      p, S_LEN, (long long)T_LEN * S_LEN,
      encT, S_LEN, (long long)H_DIM * S_LEN,
      catA, 2 * H_DIM, (long long)T_LEN * 2 * H_DIM,
      S_LEN, 1, nullptr);

  // 5) out = tanh([c|q] @ W_out^T + b)   M=BT  N=H  K=2H   (Bt = W_out itself)
  k_gemm<<<dim3(H_DIM / BN, BT / BM, 1), blk, 0, stream>>>(
      catA, 2 * H_DIM, 0,
      WoutB, 2 * H_DIM, 0,
      d_out, H_DIM, 0,
      2 * H_DIM, 2, bout);
}
